// MoxELayer_6416681140790
// MI455X (gfx1250) — matrix operand order repaired and checked by reading, not compiled, not run on hardware
//
#include <hip/hip_runtime.h>

// ---------------------------------------------------------------------------
// MoE layer for MI455X (gfx1250): bf16 WMMA expert FFN with async-to-LDS
// double-buffered staging. Shapes fixed by reference: N=8192 tokens, D=512,
// F=2048, E=8, top-2 routing.
// ---------------------------------------------------------------------------

#define USE_ASYNC_COPY 1

typedef __attribute__((ext_vector_type(16))) __bf16 v16bf;
typedef __attribute__((ext_vector_type(8)))  float v8f;

#define NTOK   8192
#define DDIM   512
#define FDIM   2048
#define NEXP   8

// d_out layout (fp32, reference tuple flattened):
#define OUT_ELEMS   (NTOK * DDIM)
#define OFF_LOGITS  (OUT_ELEMS)
#define OFF_PROBS   (OFF_LOGITS + NTOK*NEXP)
#define OFF_ZL      (OFF_PROBS  + NTOK*NEXP)
#define OFF_LB      (OFF_ZL + 1)
#define OFF_LOAD    (OFF_LB + 1)
#define OFF_CNT     (OFF_LOAD + NEXP)

struct U4 { unsigned int x, y, z, w; };
union Frag { v16bf v; U4 q[2]; };

__device__ __forceinline__ unsigned short f2bf(float f) {
    unsigned int u = __float_as_uint(f);
    unsigned int r = u + 0x7fffu + ((u >> 16) & 1u);   // RNE
    return (unsigned short)(r >> 16);
}

__device__ __forceinline__ float gelu_tanh(float x) {
    float x3 = x * x * x;
    float t = tanhf(0.7978845608028654f * (x + 0.044715f * x3));
    return 0.5f * x * (1.0f + t);
}

#if USE_ASYNC_COPY
// Per-lane async copy of 16 bytes global -> LDS (GV mode, tracked by ASYNCcnt)
__device__ __forceinline__ void async_cp16(unsigned ldsOff,
                                           const unsigned short* gptr) {
    asm volatile("global_load_async_to_lds_b128 %0, %1, off"
                 :: "v"(ldsOff),
                    "v"((unsigned long long)(size_t)gptr)
                 : "memory");
}
__device__ __forceinline__ void wait_async_le(int n) {
    if (n == 0) asm volatile("s_wait_asynccnt 0x0" ::: "memory");
    else        asm volatile("s_wait_asynccnt 0x3" ::: "memory");
}
#endif

// ---------------------------------------------------------------------------
__global__ void zero_init(float* __restrict__ out, float* __restrict__ accums) {
    unsigned idx = blockIdx.x * 256u + threadIdx.x;
    if (idx < OUT_ELEMS) out[idx] = 0.0f;
    if (blockIdx.x == 0 && threadIdx.x < 32) accums[threadIdx.x] = 0.0f;
}

// ---------------------------------------------------------------------------
// Tiled transpose + fp32->bf16 pack: dst[e][c][r] = bf16(src[e][r][c])
// grid = (C/64, R/64, NEXP), block = 256
// ---------------------------------------------------------------------------
__global__ void pack_transpose(const float* __restrict__ src,
                               unsigned short* __restrict__ dst,
                               int R, int C) {
    __shared__ float tile[64][65];
    const unsigned tid = threadIdx.x;
    const float* s = src + (size_t)blockIdx.z * R * C;
    unsigned short* d = dst + (size_t)blockIdx.z * R * C;
    const unsigned c0 = blockIdx.x * 64u, r0 = blockIdx.y * 64u;
#pragma unroll
    for (int i = 0; i < 16; ++i) {
        unsigned l = tid + 256u * i;
        unsigned rr = l >> 6, cc = l & 63u;
        tile[rr][cc] = s[(r0 + rr) * (unsigned)C + c0 + cc];
    }
    __syncthreads();
#pragma unroll
    for (int i = 0; i < 16; ++i) {
        unsigned l = tid + 256u * i;
        unsigned cr = l >> 6, rc = l & 63u;
        d[(c0 + cr) * (unsigned)R + r0 + rc] = f2bf(tile[rc][cr]);
    }
}

// ---------------------------------------------------------------------------
// Router: logits, softmax, top-2 combine weights, loss partials, bf16 pack.
// accums: [0]=sum(lse^2)  [1..8]=sum(probs_e)  [9..16]=counts_e
// ---------------------------------------------------------------------------
__global__ void router(const float* __restrict__ tokens,
                       const float* __restrict__ gate_w,
                       float* __restrict__ out_logits,
                       float* __restrict__ out_probs,
                       float* __restrict__ dw,
                       unsigned short* __restrict__ Xb,
                       float* __restrict__ accums) {
    __shared__ float sl[NEXP];
    __shared__ float sc[NEXP];
    __shared__ float sz;
    int tid = threadIdx.x;
    if (tid < NEXP) { sl[tid] = 0.0f; sc[tid] = 0.0f; }
    if (tid == 0) sz = 0.0f;
    __syncthreads();

    unsigned n = blockIdx.x * 256u + tid;
    float l[NEXP];
#pragma unroll
    for (int e = 0; e < NEXP; ++e) l[e] = 0.0f;

    const float* trow = tokens + n * (unsigned)DDIM;
    unsigned short* xrow = Xb + n * (unsigned)DDIM;
    for (int d = 0; d < DDIM; ++d) {
        float x = trow[d];
        xrow[d] = f2bf(x);
        const float* g = gate_w + d * NEXP;
#pragma unroll
        for (int e = 0; e < NEXP; ++e) l[e] += x * g[e];
    }

    float mx = l[0];
#pragma unroll
    for (int e = 1; e < NEXP; ++e) mx = fmaxf(mx, l[e]);
    float p[NEXP]; float s = 0.0f;
#pragma unroll
    for (int e = 0; e < NEXP; ++e) { p[e] = expf(l[e] - mx); s += p[e]; }
    float inv = 1.0f / s;
#pragma unroll
    for (int e = 0; e < NEXP; ++e) p[e] *= inv;
    float lse = mx + logf(s);

#pragma unroll
    for (int e = 0; e < NEXP; ++e) {
        out_logits[n * NEXP + e] = l[e];
        out_probs [n * NEXP + e] = p[e];
    }

    int i1 = 0; float v1 = p[0];
#pragma unroll
    for (int e = 1; e < NEXP; ++e) if (p[e] > v1) { v1 = p[e]; i1 = e; }
    int i2 = (i1 == 0) ? 1 : 0; float v2 = p[i2];
#pragma unroll
    for (int e = 0; e < NEXP; ++e) if (e != i1 && p[e] > v2) { v2 = p[e]; i2 = e; }
    float wsum = v1 + v2;
    float wa = v1 / wsum, wb = v2 / wsum;
#pragma unroll
    for (int e = 0; e < NEXP; ++e)
        dw[n * NEXP + e] = (e == i1) ? wa : ((e == i2) ? wb : 0.0f);

#pragma unroll
    for (int e = 0; e < NEXP; ++e) atomicAdd(&sl[e], p[e]);
    atomicAdd(&sc[i1], 1.0f);
    atomicAdd(&sc[i2], 1.0f);
    atomicAdd(&sz, lse * lse);
    __syncthreads();
    if (tid < NEXP) {
        atomicAdd(&accums[1 + tid], sl[tid]);
        atomicAdd(&accums[9 + tid], sc[tid]);
    }
    if (tid == 0) atomicAdd(&accums[0], sz);
}

// ---------------------------------------------------------------------------
// bf16 WMMA GEMM, 64x128 block tile, 8 wave32s x (32x32), double-buffered LDS
// fed by global_load_async_to_lds_b128.
// A:  bf16 [8192, K] row-major.  Bt: bf16 [N, K] row-major (k-contiguous).
// MODE 0: outH = bf16( gelu(A@B + bias) * dw[:,e] )
// MODE 1: outC += A@B + dw[:,e]*bias
// ---------------------------------------------------------------------------
template <int MODE>
__global__ void moe_gemm(const unsigned short* __restrict__ A,
                         const unsigned short* __restrict__ Bt,
                         const float* __restrict__ bias,
                         const float* __restrict__ dw,
                         int expert, int K, int N,
                         unsigned short* __restrict__ outH,
                         float* __restrict__ outC) {
    __shared__ alignas(16) unsigned short As[2][64 * 32];
    __shared__ alignas(16) unsigned short Bs[2][128 * 32];

    const unsigned tid  = threadIdx.x;
    const unsigned wave = tid >> 5;
    const unsigned lane = tid & 31u;
    const unsigned h    = lane >> 4;
    const unsigned ln   = lane & 15u;
    const unsigned wm   = wave >> 2;     // 0..1
    const unsigned wn   = wave & 3u;     // 0..3

    const unsigned m0 = blockIdx.y * 64u;
    const unsigned n0 = blockIdx.x * 128u;
    const unsigned uK = (unsigned)K, uN = (unsigned)N;

    // staging assignment (16B chunks)
    const unsigned arow = tid >> 2;             // 0..63
    const unsigned acol = (tid & 3u) * 8u;      // 0,8,16,24
    const unsigned brow = tid >> 1;             // 0..127
    const unsigned bcol = (tid & 1u) * 16u;     // 0 or 16

    const unsigned short* aSrc = A  + (m0 + arow) * uK + acol;
    const unsigned short* bSrc = Bt + (n0 + brow) * uK + bcol;

#if USE_ASYNC_COPY
    const unsigned aOff = ((unsigned)(size_t)&As[0][0]) + (arow * 32u + acol) * 2u;
    const unsigned bOff = ((unsigned)(size_t)&Bs[0][0]) + (brow * 32u + bcol) * 2u;
#endif

    v8f acc[2][2];
    {
        v8f z;
#pragma unroll
        for (int r = 0; r < 8; ++r) z[r] = 0.0f;
#pragma unroll
        for (int i = 0; i < 2; ++i)
#pragma unroll
            for (int j = 0; j < 2; ++j) acc[i][j] = z;
    }

#if USE_ASYNC_COPY
    // prologue: fill buffer 0
    async_cp16(aOff,       aSrc);
    async_cp16(bOff,       bSrc);
    async_cp16(bOff + 16u, bSrc + 8);
#endif

    unsigned it = 0;
    for (int k0 = 0; k0 < K; k0 += 32, ++it) {
        const unsigned cur = it & 1u;
        const bool hasNext = (k0 + 32 < K);
#if USE_ASYNC_COPY
        if (hasNext) {
            const unsigned nb = (cur ^ 1u);
            async_cp16(aOff + nb * 4096u,       aSrc + k0 + 32);
            async_cp16(bOff + nb * 8192u,       bSrc + k0 + 32);
            async_cp16(bOff + nb * 8192u + 16u, bSrc + k0 + 40);
        }
        wait_async_le(hasNext ? 3 : 0);
#else
        *(U4*)&As[cur][arow * 32u + acol] = *(const U4*)(aSrc + k0);
        *(U4*)&Bs[cur][brow * 32u + bcol] = *(const U4*)(bSrc + k0);
        *(U4*)&Bs[cur][brow * 32u + bcol + 8u] = *(const U4*)(bSrc + k0 + 8);
        (void)hasNext;
#endif
        __syncthreads();

        // A fragment: lane half h holds K in [8h,8h+8) and [16+8h,24+8h)
        Frag a[2];
#pragma unroll
        for (int i = 0; i < 2; ++i) {
            const unsigned short* pr = &As[cur][(wm * 32u + i * 16u + ln) * 32u];
            a[i].q[0] = *(const U4*)(pr + 8u * h);
            a[i].q[1] = *(const U4*)(pr + 16u + 8u * h);
        }
        // B fragment: lane half h holds K in [8h,8h+8) and [16+8h,24+8h), column ln
        Frag b[2];
#pragma unroll
        for (int j = 0; j < 2; ++j) {
            const unsigned short* pr =
                &Bs[cur][(wn * 32u + j * 16u + ln) * 32u];
            b[j].q[0] = *(const U4*)(pr + 8u * h);
            b[j].q[1] = *(const U4*)(pr + 16u + 8u * h);
        }
#pragma unroll
        for (int i = 0; i < 2; ++i)
#pragma unroll
            for (int j = 0; j < 2; ++j)
                acc[i][j] = __builtin_amdgcn_wmma_f32_16x16x32_bf16(
                    false, a[i].v, false, b[j].v, (short)0, acc[i][j],
                    false, false);
        __syncthreads();
    }

    // epilogue: C element (r,lane) -> m = r + 8h, n = ln within tile
#pragma unroll
    for (int i = 0; i < 2; ++i) {
#pragma unroll
        for (int j = 0; j < 2; ++j) {
#pragma unroll
            for (int r = 0; r < 8; ++r) {
                unsigned gm = m0 + wm * 32u + i * 16u + r + 8u * h;
                unsigned gn = n0 + wn * 32u + j * 16u + ln;
                float v = acc[i][j][r];
                float wv = dw[gm * NEXP + expert];
                if (MODE == 0) {
                    float g = gelu_tanh(v + bias[gn]);
                    outH[gm * uN + gn] = f2bf(g * wv);
                } else {
                    unsigned o = gm * uN + gn;
                    outC[o] += v + wv * bias[gn];
                }
            }
        }
    }
}

// ---------------------------------------------------------------------------
__global__ void finalize(const float* __restrict__ accums, float* __restrict__ out) {
    int tid = threadIdx.x;
    if (tid < NEXP) {
        out[OFF_LOAD + tid] = accums[1 + tid] / (float)NTOK;
        out[OFF_CNT + tid]  = accums[9 + tid];
    }
    if (tid == 0) {
        out[OFF_ZL] = accums[0] / (float)NTOK;
        float s = 0.0f;
#pragma unroll
        for (int e = 0; e < NEXP; ++e) {
            float frac = accums[9 + e] / (float)(NTOK * 2);
            float load = accums[1 + e] / (float)NTOK;
            s += frac * load;
        }
        out[OFF_LB] = (float)NEXP * s;
    }
}

// ---------------------------------------------------------------------------
extern "C" void kernel_launch(void* const* d_in, const int* in_sizes, int n_in,
                              void* d_out, int out_size, void* d_ws, size_t ws_size,
                              hipStream_t stream) {
    const float* h_t    = (const float*)d_in[0];
    const float* gate_w = (const float*)d_in[1];
    const float* w1     = (const float*)d_in[2];
    const float* b1     = (const float*)d_in[3];
    const float* w2     = (const float*)d_in[4];
    const float* b2     = (const float*)d_in[5];
    float* out = (float*)d_out;

    // workspace layout (~75.8 MB total)
    char* ws = (char*)d_ws;
    float* accums      = (float*)ws;                           // 64 floats
    float* dwp         = (float*)(ws + 4096);                  // [N,8] fp32
    unsigned short* Xb = (unsigned short*)(ws + 266240);       // [N,D]  bf16
    unsigned short* Hb = (unsigned short*)(ws + 8654848);      // [N,F]  bf16
    unsigned short* W1T = (unsigned short*)(ws + 42209280);    // [E,F,D] bf16
    unsigned short* W2T = (unsigned short*)(ws + 58986496);    // [E,D,F] bf16

    zero_init<<<OUT_ELEMS / 256, 256, 0, stream>>>(out, accums);
    // one-shot weight transpose+pack to k-contiguous bf16
    pack_transpose<<<dim3(FDIM / 64, DDIM / 64, NEXP), 256, 0, stream>>>(
        w1, W1T, DDIM, FDIM);
    pack_transpose<<<dim3(DDIM / 64, FDIM / 64, NEXP), 256, 0, stream>>>(
        w2, W2T, FDIM, DDIM);
    router<<<NTOK / 256, 256, 0, stream>>>(h_t, gate_w,
                                           out + OFF_LOGITS, out + OFF_PROBS,
                                           dwp, Xb, accums);
    for (int e = 0; e < NEXP; ++e) {
        moe_gemm<0><<<dim3(FDIM / 128, NTOK / 64), 256, 0, stream>>>(
            Xb, W1T + (size_t)e * FDIM * DDIM, b1 + (size_t)e * FDIM,
            dwp, e, DDIM, FDIM, Hb, nullptr);
        moe_gemm<1><<<dim3(DDIM / 128, NTOK / 64), 256, 0, stream>>>(
            Hb, W2T + (size_t)e * DDIM * FDIM, b2 + (size_t)e * DDIM,
            dwp, e, FDIM, DDIM, nullptr, out);
    }
    finalize<<<1, 32, 0, stream>>>(accums, out);
}
